// SimulatedQuantLinear_3977139716384
// MI455X (gfx1250) — matrix operand order repaired and checked by reading, not compiled, not run on hardware
//
#include <hip/hip_runtime.h>
#include <hip/hip_bf16.h>

// SimulatedQuantLinear as a bf16-WMMA GEMM on gfx1250 (wave32).
// M=8192 (4*2048), N=4096, K=4096, GROUP_SIZE=128.
// Compute-bound (275 GFLOP vs ~14us of HBM traffic at 23.3 TB/s)
// -> v_wmma_f32_16x16x32_bf16 with f32 accumulate.

typedef __attribute__((ext_vector_type(16))) __bf16  v16bf;
typedef __attribute__((ext_vector_type(8)))  __bf16  v8bf;
typedef __attribute__((ext_vector_type(2)))  __bf16  v2bf;
typedef __attribute__((ext_vector_type(8)))  float   v8f;
typedef __attribute__((ext_vector_type(4)))  float   f32x4;
typedef __attribute__((ext_vector_type(4)))  unsigned uintx4;

union frag16 { v16bf v; v8bf h[2]; };

// One v_cvt_pk_bf16_f32 per pair (instead of clang's 2x cvt + merge for scalar casts).
static __device__ __forceinline__ unsigned pkbf(float lo, float hi) {
#if __has_builtin(__builtin_amdgcn_cvt_pk_bf16_f32)
  v2bf r = __builtin_amdgcn_cvt_pk_bf16_f32(lo, hi);
  return __builtin_bit_cast(unsigned, r);
#else
  unsigned short a = __builtin_bit_cast(unsigned short, (__bf16)lo);
  unsigned short b = __builtin_bit_cast(unsigned short, (__bf16)hi);
  return (unsigned)a | ((unsigned)b << 16);
#endif
}

#define GM   8192
#define GN   4096
#define GK   4096
#define BM   128
#define BN   128
#define BK   64
#define LDSROW 144                 // 128B of bf16 + 16B pad (bank-conflict avoidance)
#define ATILE  (128 * LDSROW)      // 18432 B
#define BUFSZ  (2 * ATILE)         // 36864 B per double-buffer slot
#define SMEMSZ (2 * BUFSZ)         // 73728 B total

__global__ __launch_bounds__(256)
void qlinear_wmma_bf16(const float* __restrict__ Ain,   // [8192, 4096]
                       const float* __restrict__ Wq,    // [4096, 4096]
                       const float* __restrict__ Sc,    // [32, 4096]
                       const float* __restrict__ Bias,  // [4096]
                       float* __restrict__ Out)         // [8192, 4096]
{
  extern __shared__ char smem[];

  const int tid  = threadIdx.x;
  const int wave = tid >> 5;
  const int lane = tid & 31;
  const int hl   = lane >> 4;      // half-wave select
  const int ll   = lane & 15;

  const int bm = blockIdx.y * BM;  // grid.y = 64 (M stripes)
  const int bn = blockIdx.x * BN;  // grid.x = 32 (N, fastest -> A-stripe L2 reuse)

  const int wm = (wave >> 1) * 32; // wave M offset within block: 0..96
  const int wn = (wave & 1) * 64;  // wave N offset within block: 0 or 64

  // Global->reg staging: 4 steps x (32 rows covered), 8 floats per row chunk.
  const int lrow   = tid >> 3;     // 0..31
  const int lchunk = tid & 7;      // 0..7 -> k offset = lchunk*8

  f32x4 aSt[4][2];
  f32x4 wSt[4][2];
  float sSt[4];

  v8f acc[2][4] = {};

  auto load_tile = [&](int t) {
    const int k0 = t * BK + lchunk * 8;
    const size_t gidx = (size_t)(t >> 1) * GN + bn;   // scale group = (t*64)/128
#pragma unroll
    for (int s = 0; s < 4; ++s) {
      const int row = s * 32 + lrow;
      const float* pa = Ain + (size_t)(bm + row) * GK + k0;
      aSt[s][0] = *(const f32x4*)pa;
      aSt[s][1] = *(const f32x4*)(pa + 4);
      const float* pw = Wq + (size_t)(bn + row) * GK + k0;
      wSt[s][0] = *(const f32x4*)pw;
      wSt[s][1] = *(const f32x4*)(pw + 4);
      sSt[s] = Sc[gidx + row];
    }
  };

  auto store_tile = [&](int p) {
    char* ab = smem + p * BUFSZ;
    char* bb = ab + ATILE;
#pragma unroll
    for (int s = 0; s < 4; ++s) {
      const int row = s * 32 + lrow;
      const float sc = sSt[s];
      uintx4 av, wv;
#pragma unroll
      for (int q = 0; q < 2; ++q) {
        const f32x4 a = aSt[s][q];
        const f32x4 w = wSt[s][q] * sc;   // vector mul -> v_pk_mul_f32
        av[q * 2 + 0] = pkbf(a[0], a[1]);
        av[q * 2 + 1] = pkbf(a[2], a[3]);
        wv[q * 2 + 0] = pkbf(w[0], w[1]);
        wv[q * 2 + 1] = pkbf(w[2], w[3]);
      }
      *(uintx4*)(ab + row * LDSROW + lchunk * 16) = av;
      *(uintx4*)(bb + row * LDSROW + lchunk * 16) = wv;
    }
  };

  auto compute = [&](int p) {
    const char* ab = smem + p * BUFSZ;
    const char* bb = ab + ATILE;
#pragma unroll
    for (int kk = 0; kk < BK; kk += 32) {
      frag16 af[2], bf_[4];
#pragma unroll
      for (int i = 0; i < 2; ++i) {
        // A 16x32 bf16: lane ll holds row M=ll; K chunk [hl*8, hl*8+8) and +16.
        const char* pa = ab + (wm + i * 16 + ll) * LDSROW + (kk + hl * 8) * 2;
        af[i].h[0] = *(const v8bf*)pa;
        af[i].h[1] = *(const v8bf*)(pa + 32);
      }
#pragma unroll
      for (int j = 0; j < 4; ++j) {
        // B 32x16 bf16: lane ll holds col N=ll; K chunk [hl*8, hl*8+8) and +16.
        const char* pb = bb + (wn + j * 16 + ll) * LDSROW + (kk + hl * 8) * 2;
        bf_[j].h[0] = *(const v8bf*)pb;
        bf_[j].h[1] = *(const v8bf*)(pb + 32);
      }
#pragma unroll
      for (int i = 0; i < 2; ++i)
#pragma unroll
        for (int j = 0; j < 4; ++j)
          acc[i][j] = __builtin_amdgcn_wmma_f32_16x16x32_bf16(
              false, af[i].v, false, bf_[j].v, (short)0, acc[i][j], false, false);
    }
  };

  // Software pipeline: global loads for t+1 overlap WMMA on tile t.
  load_tile(0);
  store_tile(0);

  const int T = GK / BK;           // 64
  for (int t = 0; t < T; ++t) {
    __syncthreads();               // stores to buf (t&1) visible to all waves
    if (t + 1 < T) load_tile(t + 1);
    compute(t & 1);
    if (t + 1 < T) store_tile((t + 1) & 1);
  }

  // Epilogue: C/D 16x16 f32 layout -> VGPR r holds M = r + 8*hl, N = ll.
#pragma unroll
  for (int j = 0; j < 4; ++j) {
    const int ncol = bn + wn + j * 16 + ll;
    const float bv = Bias[ncol];
#pragma unroll
    for (int i = 0; i < 2; ++i) {
      const size_t mrow = (size_t)(bm + wm + i * 16 + hl * 8);
      float* po = Out + mrow * GN + ncol;
#pragma unroll
      for (int r = 0; r < 8; ++r)
        po[(size_t)r * GN] = acc[i][j][r] + bv;
    }
  }
}

extern "C" void kernel_launch(void* const* d_in, const int* in_sizes, int n_in,
                              void* d_out, int out_size, void* d_ws, size_t ws_size,
                              hipStream_t stream) {
  (void)in_sizes; (void)n_in; (void)out_size; (void)d_ws; (void)ws_size;
  const float* input   = (const float*)d_in[0];
  const float* qweight = (const float*)d_in[1];
  const float* scales  = (const float*)d_in[2];
  const float* bias    = (const float*)d_in[3];
  float* out = (float*)d_out;

  dim3 grid(GN / BN, GM / BM);     // (32, 64), N fastest for L2 reuse of A stripe
  dim3 block(256);
  qlinear_wmma_bf16<<<grid, block, SMEMSZ, stream>>>(input, qweight, scales, bias, out);
}
